// min_model_85830626443601
// MI455X (gfx1250) — compile-verified, matrix operand order checked by reading, not run on hardware
//
#include <hip/hip_runtime.h>

// ---------------------------------------------------------------------------
// B=16, L=2048, V=64.
//   attn1[t,s] = ew[t-s]/Z[t]  (ew=exp(v_emb), Z=prefix sum -> exact softmax)
//   l1 = Toeplitz(ew) @ onehot(idx) / Z          (WMMA, f16 in / f32 acc)
//   scores = gather(Wq^T, idx) @ l1^T            (flash-style, causal)
//   l2 = (exp(scores) @ onehot(idx)) / rowsum * Wv[0,0]
// Unshifted softmax is exact here (|score| <~ 0.1); E one-hot over all 64
// bins => rowsum(P) == rowsum(D), so no online-softmax state is needed.
// Operand tiles are precomputed in WMMA register layout; the hot loops are
// register-double-buffered global_load_b128 + v_wmma_f32_16x16x32_f16.
// ---------------------------------------------------------------------------

typedef __attribute__((ext_vector_type(16))) _Float16 v16h;
typedef __attribute__((ext_vector_type(8)))  _Float16 v8h;
typedef __attribute__((ext_vector_type(8)))  float    v8f;

#define Bn 16
#define Ln 2048
#define Vn 64
#define NT (Ln / 16)   // 128 t-tiles per batch
#define NK (Ln / 32)   // 64  k-tiles per batch

// workspace byte offsets (total ~8.6 MB)
#define EWH_OFF  0                    // 2048 f16                   (4 KB)
#define Z_OFF    4096                 // 2048 f32                   (8 KB)
#define WQT_OFF  12288                // 64x64 f16 (Wq^T)           (8 KB)
#define ATIL_OFF 20480                // 128 tiles * 32 * 16 f16    (128 KB)
#define ETIL_OFF 155648               // 16*64*4 tiles * 32*16 f16  (4 MB)
#define L1_OFF   4349952              // 16*2048*64 f16             (4 MB)

// f16 A/B operand lane->K mapping for 16x16x32 (ISA 7.12.2):
// lanes 0-15: h0..7 -> K0..7, h8..15 -> K16..23 ; lanes 16-31: +8
__device__ __forceinline__ int kmap(int hi, int h) {
  return (h < 8 ? h : h + 8) + (hi ? 8 : 0);
}

#define WMMA_F16(A, Bm, C) \
  __builtin_amdgcn_wmma_f32_16x16x32_f16(false, (A), false, (Bm), (short)0, (C), false, false)

// two aligned b128 loads -> one 16-half operand
__device__ __forceinline__ v16h ld16h(const _Float16* p) {
  v8h a = *(const v8h*)p;
  v8h b = *(const v8h*)(p + 8);
  return __builtin_shufflevector(a, b, 0,1,2,3,4,5,6,7,8,9,10,11,12,13,14,15);
}
// gather lane's B-operand halves from a row-major 64-wide f16 row
__device__ __forceinline__ v16h ld16h_row(const _Float16* row, int hi) {
  v8h a = *(const v8h*)(row + 8 * hi);
  v8h b = *(const v8h*)(row + 16 + 8 * hi);
  return __builtin_shufflevector(a, b, 0,1,2,3,4,5,6,7,8,9,10,11,12,13,14,15);
}
__device__ __forceinline__ void st16h(_Float16* p, v16h v) {
  *(v8h*)p       = __builtin_shufflevector(v, v, 0,1,2,3,4,5,6,7);
  *(v8h*)(p + 8) = __builtin_shufflevector(v, v, 8,9,10,11,12,13,14,15);
}

// ---------------------------------------------------------------------------
// Kernel 1: ew=exp(v_emb) f16, Z=inclusive prefix sum f32, Wqt = Wq^T f16.
// ---------------------------------------------------------------------------
__global__ void prep_kernel(const float* __restrict__ v_emb,
                            const float* __restrict__ Wq_w,
                            char* __restrict__ ws) {
  _Float16* ewh = (_Float16*)(ws + EWH_OFF);
  float*    Z   = (float*)(ws + Z_OFF);
  _Float16* wqt = (_Float16*)(ws + WQT_OFF);

  __shared__ float part[256];
  const int tid = threadIdx.x;

  float loc[8];
  float run = 0.f;
#pragma unroll
  for (int j = 0; j < 8; ++j) {
    float e = __expf(v_emb[tid * 8 + j]);
    ewh[tid * 8 + j] = (_Float16)e;
    run += e;
    loc[j] = run;
  }
  part[tid] = run;
  __syncthreads();
  for (int off = 1; off < 256; off <<= 1) {
    float v = (tid >= off) ? part[tid - off] : 0.f;
    __syncthreads();
    part[tid] += v;
    __syncthreads();
  }
  float excl = part[tid] - run;
#pragma unroll
  for (int j = 0; j < 8; ++j) Z[tid * 8 + j] = excl + loc[j];

  for (int e = tid; e < Vn * Vn; e += 256) {
    int v = e >> 6, u = e & 63;
    wqt[e] = (_Float16)Wq_w[u * Vn + v];
  }
}

// ---------------------------------------------------------------------------
// Kernel 1b: 128 causal-Toeplitz A-tiles in register layout (shared by all
// batches).  Tile dt: A[m,k] = ew[16*dt + m - k]  (0 if negative).
// ---------------------------------------------------------------------------
__global__ void atile_kernel(char* __restrict__ ws) {
  const _Float16* ewh = (const _Float16*)(ws + EWH_OFF);
  _Float16* atil = (_Float16*)(ws + ATIL_OFF);
  const int dt = blockIdx.x;
  const int lane = threadIdx.x, laneLo = lane & 15, hi = lane >> 4;
  const _Float16 H0 = (_Float16)0.0f;
  v16h a;
#pragma unroll
  for (int h = 0; h < 16; ++h) {
    const int off = 16 * dt + laneLo - kmap(hi, h);     // <= 2047
    const _Float16 v = ewh[off < 0 ? 0 : off];          // always-safe load
    a[h] = (off >= 0) ? v : H0;
  }
  st16h(atil + (dt * 32 + lane) * 16, a);
}

// ---------------------------------------------------------------------------
// Kernel 1c: one-hot E-tiles (B-operand layout) per (b, k-tile, v-tile).
// Built ONCE, reused by both GEMM kernels.
// ---------------------------------------------------------------------------
__global__ void etile_kernel(const int* __restrict__ idx, char* __restrict__ ws) {
  _Float16* etil = (_Float16*)(ws + ETIL_OFF);
  const int blk = blockIdx.x;
  const int b = blk >> 6, kt = blk & 63;
  const int lane = threadIdx.x, laneLo = lane & 15, hi = lane >> 4;
  const _Float16 H1 = (_Float16)1.0f, H0 = (_Float16)0.0f;
  const int* idxB = idx + b * Ln + kt * 32;
  int tok[16];
#pragma unroll
  for (int h = 0; h < 16; ++h) tok[h] = idxB[kmap(hi, h)];
  _Float16* base = etil + (size_t)((b * NK + kt) * 4) * 32 * 16 + lane * 16;
#pragma unroll
  for (int vt = 0; vt < 4; ++vt) {
    v16h e;
    const int want = vt * 16 + laneLo;
#pragma unroll
    for (int h = 0; h < 16; ++h) e[h] = (tok[h] == want) ? H1 : H0;
    st16h(base + vt * 32 * 16, e);
  }
}

// ---------------------------------------------------------------------------
// Kernel 2: l1[b,t,:] = (Toeplitz @ onehot) / Z[t]  -> f16 row-major [B,L,64].
// Register-double-buffered: step k+1's 10 b128 loads are in flight while
// step k's 4 WMMAs execute.  One wave per (b, t-tile).
// ---------------------------------------------------------------------------
struct L1Tiles { v16h a, e0, e1, e2, e3; };

__device__ __forceinline__ L1Tiles l1_load(const _Float16* atil,
                                           const _Float16* eB,
                                           int tt, int kt, int lane) {
  L1Tiles t;
  const int dt = tt - 2 * kt;                         // >= 0
  t.a = ld16h(atil + (dt * 32 + lane) * 16);
  const _Float16* eb = eB + (size_t)kt * 4 * 512;
  t.e0 = ld16h(eb + 0 * 512);
  t.e1 = ld16h(eb + 1 * 512);
  t.e2 = ld16h(eb + 2 * 512);
  t.e3 = ld16h(eb + 3 * 512);
  return t;
}

__global__ void l1_kernel(char* __restrict__ ws) {
  const _Float16* atil = (const _Float16*)(ws + ATIL_OFF);
  const _Float16* etil = (const _Float16*)(ws + ETIL_OFF);
  const float*    Z    = (const float*)(ws + Z_OFF);
  _Float16*       l1h  = (_Float16*)(ws + L1_OFF);

  const int blk = blockIdx.x;
  const int b = blk >> 7, tt = blk & 127;
  const int t0 = tt << 4;
  const int lane = threadIdx.x, laneLo = lane & 15, hi = lane >> 4;
  const _Float16* eB = etil + (size_t)(b * NK * 4) * 32 * 16 + lane * 16;

  v8f acc0 = {}, acc1 = {}, acc2 = {}, acc3 = {};
  const int ktmax = (t0 + 15) >> 5;                   // inclusive
  L1Tiles cur = l1_load(atil, eB, tt, 0, lane);
  for (int kt = 0; kt <= ktmax; ++kt) {
    const int ktn = (kt + 1 <= ktmax) ? kt + 1 : ktmax;  // branch-free clamp
    L1Tiles nxt = l1_load(atil, eB, tt, ktn, lane);
    acc0 = WMMA_F16(cur.a, cur.e0, acc0);
    acc1 = WMMA_F16(cur.a, cur.e1, acc1);
    acc2 = WMMA_F16(cur.a, cur.e2, acc2);
    acc3 = WMMA_F16(cur.a, cur.e3, acc3);
    cur = nxt;
  }

#pragma unroll
  for (int r = 0; r < 8; ++r) {
    const int t = t0 + r + (hi ? 8 : 0);
    const float invZ = 1.f / Z[t];
    _Float16* dst = l1h + ((size_t)b * Ln + t) * Vn;
    dst[ 0 + laneLo] = (_Float16)(acc0[r] * invZ);
    dst[16 + laneLo] = (_Float16)(acc1[r] * invZ);
    dst[32 + laneLo] = (_Float16)(acc2[r] * invZ);
    dst[48 + laneLo] = (_Float16)(acc3[r] * invZ);
  }
}

// ---------------------------------------------------------------------------
// Kernel 3: fused attention.  One wave per (b, 16-row t-tile).
//   S = Q @ l1^T (4 WMMA / 32-wide s-step), P = exp(S) (no max shift),
//   D += P @ onehot (4 WMMA).  Only the FINAL s-step touches the causal
//   diagonal (s0_last-1 <= t0-1 for earlier steps) => masking hoisted out.
//   rowsum(D) over the 64 one-hot bins is the softmax denominator.
//   All 8 operand tiles per step are register-double-buffered.
// ---------------------------------------------------------------------------
struct AtTiles { v16h bk00, bk01, bk10, bk11, e0, e1, e2, e3; };

__device__ __forceinline__ AtTiles at_load(const _Float16* l1B,
                                           const _Float16* eB,
                                           int s0, int laneLo, int hi) {
  AtTiles t;
  const _Float16* kr0 = l1B + (size_t)(s0 + laneLo) * Vn;   // rows s0+laneLo
  const _Float16* kr1 = kr0 + 16 * Vn;                      // rows s0+16+laneLo
  t.bk00 = ld16h_row(kr0,      hi);
  t.bk01 = ld16h_row(kr0 + 32, hi);
  t.bk10 = ld16h_row(kr1,      hi);
  t.bk11 = ld16h_row(kr1 + 32, hi);
  const _Float16* eb = eB + (size_t)(s0 >> 5) * 4 * 512;
  t.e0 = ld16h(eb + 0 * 512);
  t.e1 = ld16h(eb + 1 * 512);
  t.e2 = ld16h(eb + 2 * 512);
  t.e3 = ld16h(eb + 3 * 512);
  return t;
}

__global__ void attn_kernel(const int* __restrict__ idx,
                            const float* __restrict__ Wv_w,
                            const char* __restrict__ ws,
                            float* __restrict__ out) {
  const _Float16* l1h  = (const _Float16*)(ws + L1_OFF);
  const _Float16* wqt  = (const _Float16*)(ws + WQT_OFF);
  const _Float16* etil = (const _Float16*)(ws + ETIL_OFF);

  __shared__ alignas(16) _Float16 P[16 * 32];

  const int blk = blockIdx.x;
  const int b = blk >> 7, tt = blk & 127;
  const int t0 = tt << 4;
  const int lane = threadIdx.x, laneLo = lane & 15, hi = lane >> 4;

  // Q A-operand: row t -> row idx[b,t] of Wq^T (contiguous b128 loads)
  const int tokR = idx[b * Ln + t0 + laneLo];
  const v16h aq0 = ld16h_row(wqt + tokR * Vn, hi);
  const v16h aq1 = ld16h_row(wqt + tokR * Vn + 32, hi);

  v8f d0 = {}, d1 = {}, d2 = {}, d3 = {};

  const int s0_last = ((t0 + 15) >> 5) << 5;
  const _Float16* l1B = l1h + (size_t)b * Ln * Vn;
  const _Float16* eB  = etil + (size_t)(b * NK * 4) * 32 * 16 + lane * 16;

  AtTiles cur = at_load(l1B, eB, 0, laneLo, hi);
  for (int s0 = 0; s0 <= s0_last; s0 += 32) {
    const int s0n = (s0 + 32 <= s0_last) ? s0 + 32 : s0_last;  // clamp
    AtTiles nxt = at_load(l1B, eB, s0n, laneLo, hi);
    const bool masked = (s0 == s0_last);

    v8f c0 = {}, c1 = {};
    c0 = WMMA_F16(aq0, cur.bk00, c0);
    c0 = WMMA_F16(aq1, cur.bk01, c0);
    c1 = WMMA_F16(aq0, cur.bk10, c1);
    c1 = WMMA_F16(aq1, cur.bk11, c1);

#pragma unroll
    for (int r = 0; r < 8; ++r) {
      const int prow = r + (hi ? 8 : 0);
      float p0 = __expf(c0[r]);
      float p1 = __expf(c1[r]);
      if (masked) {
        const int trow = t0 + prow;
        p0 = (s0 + laneLo      <= trow) ? p0 : 0.f;
        p1 = (s0 + 16 + laneLo <= trow) ? p1 : 0.f;
      }
      P[prow * 32 + laneLo]      = (_Float16)p0;
      P[prow * 32 + 16 + laneLo] = (_Float16)p1;
    }
    __syncthreads();                       // 1-wave WG: ~free
    // A-layout read-back: two contiguous 8-half runs -> 2x ds_load_b128
    const v8h plo = *(const v8h*)&P[laneLo * 32 + 8 * hi];
    const v8h phi = *(const v8h*)&P[laneLo * 32 + 16 + 8 * hi];
    const v16h pa = __builtin_shufflevector(plo, phi,
        0,1,2,3,4,5,6,7,8,9,10,11,12,13,14,15);
    __syncthreads();

    d0 = WMMA_F16(pa, cur.e0, d0);
    d1 = WMMA_F16(pa, cur.e1, d1);
    d2 = WMMA_F16(pa, cur.e2, d2);
    d3 = WMMA_F16(pa, cur.e3, d3);
    cur = nxt;
  }

  // softmax denominator = rowsum over all 64 one-hot bins = rowsum of D
  const float wv00 = Wv_w[0];
#pragma unroll
  for (int r = 0; r < 8; ++r) {
    float rs = d0[r] + d1[r] + d2[r] + d3[r];
#pragma unroll
    for (int o = 1; o < 16; o <<= 1) rs += __shfl_xor(rs, o, 32);
    const float sc = wv00 / rs;
    const int t = t0 + r + (hi ? 8 : 0);
    float* dst = out + ((size_t)b * Ln + t) * Vn;
    dst[ 0 + laneLo] = d0[r] * sc;
    dst[16 + laneLo] = d1[r] * sc;
    dst[32 + laneLo] = d2[r] * sc;
    dst[48 + laneLo] = d3[r] * sc;
  }
}

// ---------------------------------------------------------------------------
extern "C" void kernel_launch(void* const* d_in, const int* in_sizes, int n_in,
                              void* d_out, int out_size, void* d_ws, size_t ws_size,
                              hipStream_t stream) {
  const int*   idx   = (const int*)d_in[0];     // [16,2048] int32
  const float* Wq_w  = (const float*)d_in[1];   // [64,64]
  const float* Wv_w  = (const float*)d_in[2];   // [64,64] (only [0,0] used)
  const float* v_emb = (const float*)d_in[3];   // [2048,1]
  char*  ws  = (char*)d_ws;                     // ~8.6 MB used
  float* out = (float*)d_out;                   // [16,2048,64] f32

  prep_kernel <<<1,        256, 0, stream>>>(v_emb, Wq_w, ws);
  atile_kernel<<<128,       32, 0, stream>>>(ws);
  etile_kernel<<<Bn * NK,   32, 0, stream>>>(idx, ws);
  l1_kernel   <<<Bn * NT,   32, 0, stream>>>(ws);
  attn_kernel <<<Bn * NT,   32, 0, stream>>>(idx, Wv_w, ws, out);
}
